// Network_45217415692696
// MI455X (gfx1250) — matrix operand order repaired and checked by reading, not compiled, not run on hardware
//
#include <hip/hip_runtime.h>

// ---------------- bf16 helpers (raw-bits) ----------------
typedef __attribute__((ext_vector_type(16))) __bf16 v16bf;
typedef __attribute__((ext_vector_type(8)))  float  v8f;

union FragBF { v16bf v; uint4 q[2]; };

__device__ __forceinline__ unsigned short f2bf(float f) {
  unsigned u = __builtin_bit_cast(unsigned, f);
  unsigned r = u + 0x7FFFu + ((u >> 16) & 1u);   // RNE
  return (unsigned short)(r >> 16);
}
__device__ __forceinline__ float bf2f(unsigned short h) {
  unsigned u = ((unsigned)h) << 16;
  return __builtin_bit_cast(float, u);
}

// ---------------- layout conversion kernels ----------------
// fp32 NCHW (C=256) -> bf16 HWC
__global__ void nchw_to_hwc_bf16(const float* __restrict__ in,
                                 unsigned short* __restrict__ out, int HW) {
  int i = blockIdx.x * blockDim.x + threadIdx.x;     // over 256*HW
  if (i >= 256 * HW) return;
  int c = i / HW, s = i - c * HW;
  out[(size_t)s * 256 + c] = f2bf(in[i]);
}

// fp32 OIHW [256][256][3][3] -> bf16 [tap(9)][cout(256)][cin(256)]
__global__ void pack_w3x3(const float* __restrict__ w,
                          unsigned short* __restrict__ wpk) {
  int i = blockIdx.x * blockDim.x + threadIdx.x;     // over 9*256*256
  if (i >= 9 * 256 * 256) return;
  int t = i >> 16;
  int rem = i & 65535;
  int co = rem >> 8, ci = rem & 255;
  wpk[i] = f2bf(w[(((size_t)co << 8) + ci) * 9 + t]);
}

// ---------------- WMMA implicit-GEMM conv3x3 (256->256, bias+ReLU) --------
// Block = 256 threads (8 waves). Block tile: 16 spatial x 128 cout
// (wave w -> cout tile). A-panel (16 spatial x 256 cin, one tap) is staged
// in LDS via async copy, double-buffered across taps.

#define LDS_ROW 264   // 256 cin + 8 pad elems (528 B) -> bank-conflict-free

__device__ __forceinline__ void fill_tile_async(
    unsigned short* __restrict__ tileg,              // generic ptr to buffer
    const unsigned short* __restrict__ in,
    int H, int W, int HW, int tile0, int t) {
  int dy = t / 3 - 1, dx = t % 3 - 1;
  const uint4 Z = {0u, 0u, 0u, 0u};
#pragma unroll
  for (int c = 0; c < 2; ++c) {
    int chunk = threadIdx.x * 2 + c;                 // 0..511
    int m = chunk >> 5;                              // spatial row 0..15
    int ci = (chunk & 31) * 8;                       // cin base (8 bf16/16B)
    int s = tile0 + m;
    int y = s / W, x = s - y * W;
    int yy = y + dy, xx = x + dx;
    unsigned short* lp = tileg + m * LDS_ROW + ci;
    bool ok = (s < HW) && (yy >= 0) && (yy < H) && (xx >= 0) && (xx < W);
    if (ok) {
      // low 32 bits of a generic LDS pointer == wave-relative LDS address
      unsigned laddr = (unsigned)(size_t)lp;
      unsigned long long gaddr =
          (unsigned long long)(size_t)(in + (size_t)(yy * W + xx) * 256 + ci);
      asm volatile("global_load_async_to_lds_b128 %0, %1, off"
                   :: "v"(laddr), "v"(gaddr) : "memory");
    } else {
      *(uint4*)lp = Z;                               // padding row -> zeros
    }
  }
}

__global__ __launch_bounds__(256)
void conv3x3_wmma(const unsigned short* __restrict__ in,   // [HW][256] bf16
                  const unsigned short* __restrict__ wpk,  // [9][256][256] bf16
                  const float* __restrict__ bias,          // [256]
                  unsigned short* __restrict__ out,        // [HW][256] bf16
                  int H, int W, int relu) {
  __shared__ unsigned short tile[2][16 * LDS_ROW];

  const int HW   = H * W;
  const int lane = threadIdx.x & 31;
  const int wave = threadIdx.x >> 5;
  const int cout0 = (blockIdx.y * 8 + wave) * 16;          // 0..240
  const int mrow = lane & 15;                              // A: M, B: N, D: N
  const int half = lane >> 4;
  const int khalfA = half * 8;
  const int khalfB = half * 16;
  const int tile0 = blockIdx.x * 16;

  v8f acc = {0.f, 0.f, 0.f, 0.f, 0.f, 0.f, 0.f, 0.f};

  fill_tile_async(tile[0], in, H, W, HW, tile0, 0);
  asm volatile("s_wait_asynccnt 0x0" ::: "memory");
  __syncthreads();

  int buf = 0;
  for (int t = 0; t < 9; ++t) {
    if (t < 8)                                             // prefetch next tap
      fill_tile_async(tile[buf ^ 1], in, H, W, HW, tile0, t + 1);

    const unsigned short* __restrict__ arow = tile[buf] + mrow * LDS_ROW;
    const unsigned short* __restrict__ brow =
        wpk + ((size_t)t * 256 + (cout0 + mrow)) * 256;
#pragma unroll
    for (int kb = 0; kb < 256; kb += 32) {
      FragBF a, b;
      a.q[0] = *(const uint4*)(arow + kb + khalfA);        // ds_load_b128
      a.q[1] = *(const uint4*)(arow + kb + 16 + khalfA);
      b.q[0] = *(const uint4*)(brow + kb + khalfA);        // global b128
      b.q[1] = *(const uint4*)(brow + kb + 16 + khalfA);
      acc = __builtin_amdgcn_wmma_f32_16x16x32_bf16(
          false, a.v, false, b.v, (short)0, acc, false, false);
    }

    if (t < 8) {
      asm volatile("s_wait_asynccnt 0x0" ::: "memory");
      __syncthreads();
    }
    buf ^= 1;
  }

  float bia = bias[cout0 + mrow];
#pragma unroll
  for (int r = 0; r < 8; ++r) {
    int so = tile0 + r + half * 8;                         // D: M = r + 8*half
    if (so < HW) {
      float v = acc[r] + bia;
      if (relu) v = v > 0.f ? v : 0.f;
      out[(size_t)so * 256 + cout0 + mrow] = f2bf(v);
    }
  }
}

// ---------------- small head convs (256 -> Cout in {1,8}), fp32 out -------
__global__ void conv3x3_small(const unsigned short* __restrict__ in, // [HW][256]
                              const float* __restrict__ w,           // [Cout][256][9]
                              const float* __restrict__ b,
                              float* __restrict__ out,               // [HW][Cout]
                              int H, int W, int Cout) {
  int i = blockIdx.x * blockDim.x + threadIdx.x;
  int HW = H * W;
  if (i >= HW * Cout) return;
  int s = i / Cout, co = i - s * Cout;
  int y = s / W, x = s - y * W;
  float acc = b[co];
  for (int t = 0; t < 9; ++t) {
    int dy = t / 3 - 1, dx = t % 3 - 1;
    int yy = y + dy, xx = x + dx;
    if (yy < 0 || yy >= H || xx < 0 || xx >= W) continue;
    const unsigned short* arow = in + (size_t)(yy * W + xx) * 256;
    const float* wr = w + (size_t)co * 256 * 9 + t;
    for (int ci = 0; ci < 256; ++ci) acc += bf2f(arow[ci]) * wr[ci * 9];
  }
  out[i] = acc;
}

// ---------------- deterministic top-k (k=1000) per level ------------------
__device__ __forceinline__ unsigned mono(float f) {
  unsigned u = __builtin_bit_cast(unsigned, f);
  return (u & 0x80000000u) ? ~u : (u | 0x80000000u);
}

__global__ __launch_bounds__(1024)
void topk_select(const float* __restrict__ score, int n, int k,
                 int* __restrict__ out_idx) {
  __shared__ unsigned hist[256];
  __shared__ unsigned sh_prefix;
  __shared__ int sh_kneed;
  __shared__ int cnt;
  __shared__ unsigned long long arr[1024];
  int tid = threadIdx.x;
  if (tid == 0) { sh_prefix = 0; sh_kneed = k; cnt = 0; }
  __syncthreads();
  for (int pass = 3; pass >= 0; --pass) {
    if (tid < 256) hist[tid] = 0;
    __syncthreads();
    unsigned prefix = sh_prefix;
    for (int i = tid; i < n; i += 1024) {
      unsigned key = mono(score[i]);
      int sh = (pass + 1) * 8;
      bool match = (pass == 3) || ((key >> sh) == (prefix >> sh));
      if (match) atomicAdd(&hist[(key >> (pass * 8)) & 255], 1u);
    }
    __syncthreads();
    if (tid == 0) {
      int need = sh_kneed;
      unsigned d = 0;
      for (int di = 255; di >= 0; --di) {
        int c = (int)hist[di];
        if (need <= c) { d = (unsigned)di; break; }
        need -= c;
      }
      sh_kneed = need;
      sh_prefix = prefix | (d << (pass * 8));
    }
    __syncthreads();
  }
  unsigned T = sh_prefix;
  int neq = sh_kneed;
  for (int i = tid; i < n; i += 1024) {
    unsigned key = mono(score[i]);
    if (key > T) {
      int p = atomicAdd(&cnt, 1);
      if (p < k) arr[p] = (((unsigned long long)(~key)) << 32) | (unsigned)i;
    }
  }
  __syncthreads();
  if (tid == 0) {                      // ties: lowest indices, deterministic
    int base = cnt, taken = 0;
    for (int i = 0; i < n && taken < neq; ++i) {
      unsigned key = mono(score[i]);
      if (key == T) {
        arr[base + taken] = (((unsigned long long)(~key)) << 32) | (unsigned)i;
        ++taken;
      }
    }
  }
  __syncthreads();
  if (tid >= k) arr[tid] = ~0ull;
  __syncthreads();
  for (int kk = 2; kk <= 1024; kk <<= 1) {
    for (int j = kk >> 1; j > 0; j >>= 1) {
      int ixj = tid ^ j;
      if (ixj > tid) {
        unsigned long long a = arr[tid], b = arr[ixj];
        bool up = ((tid & kk) == 0);
        if ((a > b) == up) { arr[tid] = b; arr[ixj] = a; }
      }
      __syncthreads();
    }
  }
  if (tid < k) out_idx[tid] = (int)(arr[tid] & 0xFFFFFFFFu);
}

// ---------------- final decode: anchors + sigmoid + bbox_transform_inv ----
__global__ void emit_rows(const float* __restrict__ scores,
                          const float* __restrict__ regs,   // [total][8]
                          const int* __restrict__ idxl,     // [3][1024]
                          float* __restrict__ out) {
  const int LW5[5]  = {224, 112, 56, 28, 14};
  const int LSTR[5] = {8, 16, 32, 64, 128};
  const int ROW0[5] = {0, 1000, 2000, 3000, 3448};
  const int SOFF[5] = {0, 28672, 35840, 37632, 38080};
  int r = blockIdx.x * blockDim.x + threadIdx.x;
  if (r >= 3560) return;
  int lvl = 4;
  if (r < 1000) lvl = 0; else if (r < 2000) lvl = 1;
  else if (r < 3000) lvl = 2; else if (r < 3448) lvl = 3;
  int rank = r - ROW0[lvl];
  int idx = (lvl < 3) ? idxl[lvl * 1024 + rank] : rank;
  int Wl = LW5[lvl];
  float st = (float)LSTR[lvl];
  int y = idx / Wl, x = idx - y * Wl;
  float cxa = (x + 0.5f) * st, cya = (y + 0.5f) * st;
  float sh = 0.5f * 4.0f * st;
  float x1 = cxa - sh, y1 = cya - sh, x2 = cxa + sh, y2 = cya + sh;
  float w = x2 - x1 + 1.f, h = y2 - y1 + 1.f;
  float cx = x1 + 0.5f * w, cy = y1 + 0.5f * h;
  const float* d = regs + (size_t)(SOFF[lvl] + idx) * 8;
  float pcx = d[0] * w + cx, pcy = d[1] * h + cy;
  float pw = expf(d[2]) * w, ph = expf(d[3]) * h;
  float sc = scores[SOFF[lvl] + idx];
  float sig = 1.f / (1.f + expf(-sc));
  float* o = out + (size_t)r * 6;
  o[0] = pcx - 0.5f * pw; o[1] = pcy - 0.5f * ph;
  o[2] = pcx + 0.5f * pw; o[3] = pcy + 0.5f * ph;
  o[4] = sig; o[5] = 1.0f;
}

// ---------------- host orchestration --------------------------------------
extern "C" void kernel_launch(void* const* d_in, const int* in_sizes, int n_in,
                              void* d_out, int out_size, void* d_ws,
                              size_t ws_size, hipStream_t stream) {
  (void)out_size; (void)ws_size;
  const float* feat[5];
  for (int i = 0; i < 5; ++i) feat[i] = (const float*)d_in[i];

  // Identify params by size; sorted-key flatten order:
  // bbox_pred, bbox_subnet(x4), cls_score, cls_subnet(x4), refined(dead).
  const float* sub_w[8] = {};
  const float* sub_b[8] = {};
  int nsw = 0, nsb = 0;
  const float *pred8_w = nullptr, *pred8_b = nullptr;
  const float *cls1_w = nullptr, *cls1_b = nullptr;
  for (int i = 5; i < n_in; ++i) {
    int sz = in_sizes[i];
    const float* p = (const float*)d_in[i];
    if (sz == 589824 && nsw < 8) sub_w[nsw++] = p;        // 0..3 bbox, 4..7 cls
    else if (sz == 256 && nsb < 8) sub_b[nsb++] = p;
    else if (sz == 18432 && !pred8_w) pred8_w = p;
    else if (sz == 8 && !pred8_b) pred8_b = p;
    else if (sz == 2304 && !cls1_w) cls1_w = p;
    else if (sz == 1 && !cls1_b) cls1_b = p;
  }

  char* wp = (char*)d_ws;
  auto carve = [&](size_t bytes) -> void* {
    void* r = (void*)wp;
    wp += (bytes + 255) & ~(size_t)255;
    return r;
  };
  const size_t WPK_ELEMS = (size_t)9 * 256 * 256;
  const size_t ACT_ELEMS = (size_t)28672 * 256;
  unsigned short* wpk  = (unsigned short*)carve(8 * WPK_ELEMS * 2);
  unsigned short* actA = (unsigned short*)carve(ACT_ELEMS * 2);
  unsigned short* actB = (unsigned short*)carve(ACT_ELEMS * 2);
  float* scores = (float*)carve((size_t)38192 * 4);
  float* regs   = (float*)carve((size_t)38192 * 8 * 4);
  int*   idxl   = (int*)carve((size_t)3 * 1024 * 4);

  for (int i = 0; i < 8; ++i)
    pack_w3x3<<<(9 * 256 * 256 + 255) / 256, 256, 0, stream>>>(
        sub_w[i], wpk + (size_t)i * WPK_ELEMS);

  static const int LH[5] = {128, 64, 32, 16, 8};
  static const int LW5[5] = {224, 112, 56, 28, 14};
  static const int LN[5] = {28672, 7168, 1792, 448, 112};
  static const int SOFF[5] = {0, 28672, 35840, 37632, 38080};

  for (int l = 0; l < 5; ++l) {
    int H = LH[l], W = LW5[l], HW = LN[l];
    dim3 cg((HW + 15) / 16, 2), cb(256);
    int cvt = (256 * HW + 255) / 256;

    // cls branch (subnets 4..7)
    nchw_to_hwc_bf16<<<cvt, 256, 0, stream>>>(feat[l], actA, HW);
    conv3x3_wmma<<<cg, cb, 0, stream>>>(actA, wpk + 4 * WPK_ELEMS, sub_b[4], actB, H, W, 1);
    conv3x3_wmma<<<cg, cb, 0, stream>>>(actB, wpk + 5 * WPK_ELEMS, sub_b[5], actA, H, W, 1);
    conv3x3_wmma<<<cg, cb, 0, stream>>>(actA, wpk + 6 * WPK_ELEMS, sub_b[6], actB, H, W, 1);
    conv3x3_wmma<<<cg, cb, 0, stream>>>(actB, wpk + 7 * WPK_ELEMS, sub_b[7], actA, H, W, 1);
    conv3x3_small<<<(HW + 255) / 256, 256, 0, stream>>>(
        actA, cls1_w, cls1_b, scores + SOFF[l], H, W, 1);

    // bbox branch (subnets 0..3)
    nchw_to_hwc_bf16<<<cvt, 256, 0, stream>>>(feat[l], actA, HW);
    conv3x3_wmma<<<cg, cb, 0, stream>>>(actA, wpk + 0 * WPK_ELEMS, sub_b[0], actB, H, W, 1);
    conv3x3_wmma<<<cg, cb, 0, stream>>>(actB, wpk + 1 * WPK_ELEMS, sub_b[1], actA, H, W, 1);
    conv3x3_wmma<<<cg, cb, 0, stream>>>(actA, wpk + 2 * WPK_ELEMS, sub_b[2], actB, H, W, 1);
    conv3x3_wmma<<<cg, cb, 0, stream>>>(actB, wpk + 3 * WPK_ELEMS, sub_b[3], actA, H, W, 1);
    conv3x3_small<<<(HW * 8 + 255) / 256, 256, 0, stream>>>(
        actA, pred8_w, pred8_b, regs + (size_t)SOFF[l] * 8, H, W, 8);

    if (l < 3)
      topk_select<<<1, 1024, 0, stream>>>(scores + SOFF[l], HW, 1000,
                                          idxl + l * 1024);
    // 'refined' chain is dead code in the reference: skipped.
  }

  emit_rows<<<(3560 + 255) / 256, 256, 0, stream>>>(scores, regs, idxl,
                                                    (float*)d_out);
}
